// MultiheadAttention_51479478010562
// MI455X (gfx1250) — compile-verified, matrix operand order checked by reading, not run on hardware
//
#include <hip/hip_runtime.h>
#include <hip/hip_bf16.h>

// ---------------------------------------------------------------------------
// MultiheadAttention forward for MI455X (gfx1250), f16-WMMA / f32-accumulate.
//   E=512, H=8, HD=64, TGT=512, SRC=2048, B=16. SRC+bias row = 2049, padded
//   to 2080 (multiple of 32 for v_wmma_f32_16x16x32_f16 K-steps).
// Outputs (concat in d_out): out[512*16*512] f32, attn[128*512*2049] f32.
// ---------------------------------------------------------------------------

typedef __attribute__((ext_vector_type(16))) _Float16 v16h;
typedef __attribute__((ext_vector_type(8)))  _Float16 v8h;
typedef __attribute__((ext_vector_type(8)))  float    v8f;
typedef __attribute__((ext_vector_type(4)))  float    v4f;

#define E_     512
#define H_     8
#define HD_    64
#define TGT_   512
#define SRC_   2048
#define B_     16
#define SRC2_  2049
#define SRCP_  2080
#define NHEAD_ 128

__device__ __forceinline__ v8f wmma16(v16h a, v16h b, v8f c) {
  // D = A(16x32 f16) x B(32x16 f16) + C(16x16 f32)
  return __builtin_amdgcn_wmma_f32_16x16x32_f16(false, a, false, b,
                                                (short)0, c, false, false);
}

// Fragment loader for both A and B operands (per CDNA5 ISA 16-bit layouts):
// this lane's row/col = lane&15, kbase = (lane>>4)*8; element e holds
// k = kbase + (e&7) + (e>=8 ? 16 : 0). Caller passes p = &elem(row, kbase);
// we read two contiguous 8-half (16-byte) chunks at +0 and +16.
__device__ __forceinline__ v16h ld_frag_h(const _Float16* p) {
  v8h lo = *(const v8h*)(p);
  v8h hi = *(const v8h*)(p + 16);
  v16h r;
#pragma unroll
  for (int i = 0; i < 8; ++i) { r[i] = lo[i]; r[i + 8] = hi[i]; }
  return r;
}

// Same fragment pattern, but source is f32 (global or LDS); convert to f16.
__device__ __forceinline__ v16h ld_frag_f32(const float* p) {
  v4f a0 = *(const v4f*)(p);
  v4f a1 = *(const v4f*)(p + 4);
  v4f b0 = *(const v4f*)(p + 16);
  v4f b1 = *(const v4f*)(p + 20);
  v16h r;
#pragma unroll
  for (int i = 0; i < 4; ++i) {
    r[i]      = (_Float16)a0[i];
    r[i + 4]  = (_Float16)a1[i];
    r[i + 8]  = (_Float16)b0[i];
    r[i + 12] = (_Float16)b1[i];
  }
  return r;
}

// ---------------------------------------------------------------------------
__global__ __launch_bounds__(256) void cvt_f32_f16_kernel(
    const float* __restrict__ src, _Float16* __restrict__ dst, int n) {
  int i = blockIdx.x * 256 + threadIdx.x;
  if (i < n) dst[i] = (_Float16)src[i];
}

// Fill k_h / vT_h rows s=2048 (bias_k/bias_v, raw per reference) and
// s=2049..2079 (zero pad). grid=(128 heads), block=(64 hd).
__global__ __launch_bounds__(64) void pad_kv_kernel(
    const float* __restrict__ bias_k, const float* __restrict__ bias_v,
    _Float16* __restrict__ k_h, _Float16* __restrict__ vT_h) {
  int head = blockIdx.x, hd = threadIdx.x;
  int h = head & (H_ - 1);
  _Float16 bk = (_Float16)bias_k[h * HD_ + hd];
  _Float16 bv = (_Float16)bias_v[h * HD_ + hd];
  k_h[((size_t)head * SRCP_ + 2048) * HD_ + hd] = bk;
  for (int s = 2049; s < SRCP_; ++s)
    k_h[((size_t)head * SRCP_ + s) * HD_ + hd] = (_Float16)0.f;
  _Float16* vrow = vT_h + ((size_t)head * HD_ + hd) * SRCP_;
  vrow[2048] = bv;
  for (int s = 2049; s < SRCP_; ++s) vrow[s] = (_Float16)0.f;
}

// Input projections. blockIdx.y = z (0:q 1:k 2:v). Each wave computes a
// 16(M) x 64(N) strip with 4 WMMA accumulators over K=512.
__global__ __launch_bounds__(128) void proj_kernel(
    const float* __restrict__ query, const float* __restrict__ key,
    const float* __restrict__ value, const _Float16* __restrict__ Wf16,
    const float* __restrict__ in_bias, _Float16* __restrict__ q_h,
    _Float16* __restrict__ k_h, _Float16* __restrict__ vT_h) {
  const int z = blockIdx.y;
  const int gwave = blockIdx.x * 4 + (threadIdx.x >> 5);
  const int Mtiles = (z == 0) ? (TGT_ * B_ / 16) : (SRC_ * B_ / 16);
  const int mtile = gwave >> 3;   // 8 N-strips of 64
  const int nstrip = gwave & 7;
  if (mtile >= Mtiles) return;

  const float* X = (z == 0) ? query : (z == 1) ? key : value;
  const _Float16* W = Wf16 + (size_t)z * E_ * E_;

  const int lane = threadIdx.x & 31;
  const int r = lane & 15;
  const int kb = (lane >> 4) * 8;

  v8f acc[4] = {v8f{}, v8f{}, v8f{}, v8f{}};
  const float* arow = X + (size_t)(mtile * 16 + r) * E_ + kb;
  const _Float16* bbase = W + (size_t)(nstrip * 64 + r) * E_ + kb;

  for (int kk = 0; kk < E_ / 32; ++kk) {
    v16h a = ld_frag_f32(arow + kk * 32);
#pragma unroll
    for (int nt = 0; nt < 4; ++nt) {
      v16h b = ld_frag_h(bbase + (size_t)nt * 16 * E_ + kk * 32);
      acc[nt] = wmma16(a, b, acc[nt]);
    }
  }

  const int mhi = (lane >> 4) * 8;
#pragma unroll
  for (int nt = 0; nt < 4; ++nt) {
    int nf = nstrip * 64 + nt * 16 + r;       // output feature
    float bias = in_bias[z * E_ + nf];
    int head_f = nf >> 6, hd = nf & 63;
#pragma unroll
    for (int rr = 0; rr < 8; ++rr) {
      int m = mtile * 16 + rr + mhi;          // token row = t*B + b
      int bb = m & (B_ - 1), tt = m >> 4;
      int head = bb * H_ + head_f;
      float v = acc[nt][rr] + bias;
      if (z == 0) {
        v *= 0.125f;                          // HD^-0.5
        q_h[((size_t)head * TGT_ + tt) * HD_ + hd] = (_Float16)v;
      } else if (z == 1) {
        k_h[((size_t)head * SRCP_ + tt) * HD_ + hd] = (_Float16)v;
      } else {
        vT_h[((size_t)head * HD_ + hd) * SRCP_ + tt] = (_Float16)v;
      }
    }
  }
}

// Fused attention: grid=(16 tgt-blocks, 128 heads), block=256 (8 waves).
// LDS holds the full 32 x 2080 f32 score tile (266 KB of the 320 KB WGP LDS).
__global__ __launch_bounds__(256) void attn_kernel(
    const _Float16* __restrict__ q_h, const _Float16* __restrict__ k_h,
    const _Float16* __restrict__ vT_h, const unsigned char* __restrict__ kpm,
    float* __restrict__ attn_out, _Float16* __restrict__ out_h) {
  extern __shared__ float smem[];
  float* sc = smem;                       // [32][SRCP_]
  float* rowmax = smem + 32 * SRCP_;      // [32]
  float* rowsum = rowmax + 32;            // [32]

  const int head = blockIdx.y;
  const int b = head >> 3;
  const int t0 = blockIdx.x * 32;
  const int wave = threadIdx.x >> 5;
  const int lane = threadIdx.x & 31;
  const int r = lane & 15;
  const int kb = (lane >> 4) * 8;

  // Preload Q fragments for the 32 rows (2 m-tiles x 2 K-steps of 32).
  const _Float16* qbase = q_h + ((size_t)head * TGT_ + t0) * HD_;
  v16h aq[2][2];
#pragma unroll
  for (int mt = 0; mt < 2; ++mt)
#pragma unroll
    for (int kk = 0; kk < 2; ++kk)
      aq[mt][kk] = ld_frag_h(qbase + (size_t)(mt * 16 + r) * HD_ + kk * 32 + kb);

  // ---- Stage A: scores = q @ k^T into LDS (masked) ----
  for (int nt = wave; nt < SRCP_ / 16; nt += 8) {
    v8f c0 = {}, c1 = {};
    const _Float16* kbase = k_h + ((size_t)head * SRCP_ + nt * 16 + r) * HD_ + kb;
#pragma unroll
    for (int kk = 0; kk < 2; ++kk) {
      v16h bf = ld_frag_h(kbase + kk * 32);
      c0 = wmma16(aq[0][kk], bf, c0);
      c1 = wmma16(aq[1][kk], bf, c1);
    }
    int s = nt * 16 + r;  // source column this lane owns
    bool masked = (s >= SRC2_) || (s < SRC_ && kpm[(size_t)b * SRC_ + s] != 0);
    int mhi = (lane >> 4) * 8;
#pragma unroll
    for (int rr = 0; rr < 8; ++rr) {
      sc[(size_t)(rr + mhi) * SRCP_ + s]      = masked ? -1e30f : c0[rr];
      sc[(size_t)(rr + mhi + 16) * SRCP_ + s] = masked ? -1e30f : c1[rr];
    }
  }
  __syncthreads();

  // ---- Stage B1: per-row max & sum(exp) (wave handles 4 rows) ----
  for (int rr = wave * 4; rr < wave * 4 + 4; ++rr) {
    float m = -3.4e38f;
    for (int c = lane; c < SRCP_; c += 32) m = fmaxf(m, sc[(size_t)rr * SRCP_ + c]);
#pragma unroll
    for (int off = 16; off > 0; off >>= 1) m = fmaxf(m, __shfl_xor(m, off));
    float ssum = 0.f;
    for (int c = lane; c < SRCP_; c += 32)
      ssum += __expf(sc[(size_t)rr * SRCP_ + c] - m);
#pragma unroll
    for (int off = 16; off > 0; off >>= 1) ssum += __shfl_xor(ssum, off);
    if (lane == 0) { rowmax[rr] = m; rowsum[rr] = ssum; }
  }
  __syncthreads();

  // ---- Stage B2: normalize in LDS + write f32 attn to d_out ----
  const size_t attn_base = ((size_t)head * TGT_ + t0) * SRC2_;
  for (int rr = 0; rr < 32; ++rr) {
    float m = rowmax[rr], inv = 1.0f / rowsum[rr];
    for (int c = threadIdx.x; c < SRCP_; c += 256) {
      float p = __expf(sc[(size_t)rr * SRCP_ + c] - m) * inv;
      sc[(size_t)rr * SRCP_ + c] = p;
      if (c < SRC2_) attn_out[attn_base + (size_t)rr * SRC2_ + c] = p;
    }
  }
  __syncthreads();

  // ---- Stage C: out_h = attn @ v  (each wave owns one 16x16 tile) ----
  const int mt = wave >> 2;        // 0..1
  const int ntc = wave & 3;        // 0..3 (16 hd cols each)
  v8f acc = {};
  const _Float16* vb = vT_h + ((size_t)head * HD_ + ntc * 16 + r) * SRCP_ + kb;
  const float* arow = sc + (size_t)(mt * 16 + r) * SRCP_ + kb;
  for (int kk = 0; kk < SRCP_ / 32; ++kk) {
    v16h a = ld_frag_f32(arow + kk * 32);   // f32 LDS probs -> f16 fragment
    v16h bf = ld_frag_h(vb + kk * 32);
    acc = wmma16(a, bf, acc);
  }
  const int mhi = (lane >> 4) * 8;
  const int hd = ntc * 16 + r;
#pragma unroll
  for (int rr = 0; rr < 8; ++rr) {
    int t = t0 + mt * 16 + rr + mhi;
    out_h[((size_t)head * TGT_ + t) * HD_ + hd] = (_Float16)acc[rr];
  }
}

// Output projection: tokens(8192) x E @ Wo^T + bias -> d_out[:4194304].
__global__ __launch_bounds__(128) void outproj_kernel(
    const _Float16* __restrict__ out_h, const _Float16* __restrict__ Wo,
    const float* __restrict__ bias, float* __restrict__ out) {
  const int gwave = blockIdx.x * 4 + (threadIdx.x >> 5);
  const int mtile = gwave >> 3;
  const int nstrip = gwave & 7;
  const int lane = threadIdx.x & 31;
  const int r = lane & 15;
  const int kb = (lane >> 4) * 8;

  const int m = mtile * 16 + r;            // token row = t*B + b
  const int bb = m & (B_ - 1), tt = m >> 4;

  v8f acc[4] = {v8f{}, v8f{}, v8f{}, v8f{}};
  const _Float16* bbase = Wo + (size_t)(nstrip * 64 + r) * E_ + kb;
  for (int kk = 0; kk < E_ / 32; ++kk) {
    // feature k = kk*32 + (chunks); head = k/64, hd = k%64 (chunks of 8 stay
    // inside one head's 64-wide slab since 32 | 64).
    const _Float16* ap = out_h +
        (((size_t)(bb * H_ + (kk >> 1)) * TGT_ + tt) * HD_ + (kk & 1) * 32 + kb);
    v16h a = ld_frag_h(ap);
#pragma unroll
    for (int nt = 0; nt < 4; ++nt) {
      v16h bf = ld_frag_h(bbase + (size_t)nt * 16 * E_ + kk * 32);
      acc[nt] = wmma16(a, bf, acc[nt]);
    }
  }
  const int mhi = (lane >> 4) * 8;
#pragma unroll
  for (int nt = 0; nt < 4; ++nt) {
    int nf = nstrip * 64 + nt * 16 + r;
    float bv = bias[nf];
#pragma unroll
    for (int rr = 0; rr < 8; ++rr) {
      int mm = mtile * 16 + rr + mhi;
      out[(size_t)mm * E_ + nf] = acc[nt][rr] + bv;
    }
  }
}

// ---------------------------------------------------------------------------
extern "C" void kernel_launch(void* const* d_in, const int* in_sizes, int n_in,
                              void* d_out, int out_size, void* d_ws,
                              size_t ws_size, hipStream_t stream) {
  (void)in_sizes; (void)n_in; (void)out_size; (void)ws_size;
  const float* query  = (const float*)d_in[0];
  const float* key    = (const float*)d_in[1];
  const float* value  = (const float*)d_in[2];
  const float* ipw    = (const float*)d_in[3];   // (1536,512)
  const float* ipb    = (const float*)d_in[4];   // (1536,)
  const float* bias_k = (const float*)d_in[5];   // (512,)
  const float* bias_v = (const float*)d_in[6];   // (512,)
  const float* opw    = (const float*)d_in[7];   // (512,512)
  const float* opb    = (const float*)d_in[8];   // (512,)
  const unsigned char* kpm = (const unsigned char*)d_in[9];  // bool (16,2048)

  // Workspace carve (bytes, all 256-aligned). Total ~87 MB.
  char* ws = (char*)d_ws;
  _Float16* Wqkv = (_Float16*)(ws);                      // 1536*512 f16
  _Float16* Wo   = (_Float16*)(ws + 1572864);            // 512*512 f16
  _Float16* q_h  = (_Float16*)(ws + 2097152);            // 128*512*64
  _Float16* k_h  = (_Float16*)(ws + 10485760);           // 128*2080*64
  _Float16* vT_h = (_Float16*)(ws + 44564480);           // 128*64*2080
  _Float16* out_h= (_Float16*)(ws + 78643200);           // 128*512*64

  float* outF = (float*)d_out;
  float* attnF = outF + (size_t)TGT_ * B_ * E_;

  cvt_f32_f16_kernel<<<(1536 * 512 + 255) / 256, 256, 0, stream>>>(
      ipw, Wqkv, 1536 * 512);
  cvt_f32_f16_kernel<<<(512 * 512 + 255) / 256, 256, 0, stream>>>(
      opw, Wo, 512 * 512);
  pad_kv_kernel<<<NHEAD_, 64, 0, stream>>>(bias_k, bias_v, k_h, vT_h);

  // z=0:q (512 m-tiles), z=1:k, z=2:v (2048 m-tiles); 8 n-strips; 4 waves/blk.
  proj_kernel<<<dim3(4096, 3), 128, 0, stream>>>(query, key, value, Wqkv, ipb,
                                                 q_h, k_h, vT_h);

  size_t smem = (size_t)(32 * SRCP_ + 64) * sizeof(float);  // 266,496 B
  attn_kernel<<<dim3(TGT_ / 32, NHEAD_), 256, smem, stream>>>(
      q_h, k_h, vT_h, kpm, attnF, out_h);

  outproj_kernel<<<(512 * 8) / 4, 128, 0, stream>>>(out_h, Wo, opb, outF);
}
